// MultiHeadAttention_9981503996505
// MI455X (gfx1250) — matrix operand order repaired and checked by reading, not compiled, not run on hardware
//
#include <hip/hip_runtime.h>

// ---------------------------------------------------------------------------
// MHA forward on MI455X (gfx1250), compile-only target.
// f32->bf16 convert -> bf16 WMMA projection GEMMs (async-LDS staged B) ->
// flash attention (async-LDS staged K/V tiles, online softmax, WMMA QK^T/PV)
// -> bf16 WMMA output projection with fp32 result.
// CDNA5 paths used: v_wmma_f32_16x16x32_bf16, global_load_async_to_lds_b128
// (ASYNCcnt + s_wait_asynccnt), ds_swizzle row reductions, split-counter waits.
// ---------------------------------------------------------------------------

typedef unsigned short u16;
typedef unsigned int   u32;
typedef __attribute__((ext_vector_type(16))) __bf16 v16bf;
typedef __attribute__((ext_vector_type(8)))  float  v8f;
typedef __attribute__((ext_vector_type(4)))  unsigned int u32x4;

static constexpr int BB = 4;
static constexpr int SS = 2048;
static constexpr int DD = 1024;
static constexpr int HH = 16;
static constexpr int DK = 64;

union FragU { u32x4 u[2]; v16bf v; };

__device__ __forceinline__ u16 f2bf(float f) {
    u32 u = __float_as_uint(f);
    u32 r = u + 0x7FFFu + ((u >> 16) & 1u);   // RNE
    return (u16)(r >> 16);
}

// A-fragment (16-bit 16x32 MxK): lane L = row M=L%16.
// lane<16: K={0..7},{16..23}; lane>=16: K={8..15},{24..31}
__device__ __forceinline__ v16bf load_frag_A(const u16* rowptr, int lane) {
    int lo = (lane < 16) ? 0 : 8;
    FragU fu;
    fu.u[0] = *reinterpret_cast<const u32x4*>(rowptr + lo);
    fu.u[1] = *reinterpret_cast<const u32x4*>(rowptr + lo + 16);
    return fu.v;
}

// B-fragment (16-bit 32x16 KxN): lane L = col N=L%16;
// lane<16: K={0..7},{16..23}; lane>=16: K={8..15},{24..31}
__device__ __forceinline__ v16bf load_frag_B(const u16* colptr, int lane) {
    int lo = (lane < 16) ? 0 : 8;
    FragU fu;
    fu.u[0] = *reinterpret_cast<const u32x4*>(colptr + lo);
    fu.u[1] = *reinterpret_cast<const u32x4*>(colptr + lo + 16);
    return fu.v;
}

__device__ __forceinline__ v8f wmma_bf16(v16bf a, v16bf b, v8f c) {
    return __builtin_amdgcn_wmma_f32_16x16x32_bf16(
        false, a, false, b, (short)0, c, false, false);
}

// Async 16B copy global -> LDS (ASYNCcnt-tracked Tensor/async path).
__device__ __forceinline__ void async_copy_b128(void* lds_dst, const void* gsrc) {
    // Generic shared pointer low 32 bits == wave-relative LDS byte address.
    u32 lds = (u32)(unsigned long long)lds_dst;
    asm volatile("global_load_async_to_lds_b128 %0, %1, off"
                 :: "v"(lds), "v"(gsrc) : "memory");
}
__device__ __forceinline__ void wait_async0() {
    asm volatile("s_wait_asynccnt 0x0" ::: "memory");
}

// 16-lane-group reductions via ds_swizzle (xor masks 1/2/4/8, and_mask 0x1F).
template <int PAT>
__device__ __forceinline__ float ds_swz(float x) {
    return __int_as_float(__builtin_amdgcn_ds_swizzle(__float_as_int(x), PAT));
}
__device__ __forceinline__ float red16_max(float v) {
    v = fmaxf(v, ds_swz<0x041F>(v));
    v = fmaxf(v, ds_swz<0x081F>(v));
    v = fmaxf(v, ds_swz<0x101F>(v));
    v = fmaxf(v, ds_swz<0x201F>(v));
    return v;
}
__device__ __forceinline__ float red16_sum(float v) {
    v += ds_swz<0x041F>(v);
    v += ds_swz<0x081F>(v);
    v += ds_swz<0x101F>(v);
    v += ds_swz<0x201F>(v);
    return v;
}

// ---------------------------------------------------------------------------
// Kernel 1: fp32 -> bf16 elementwise
// ---------------------------------------------------------------------------
__global__ void f32_to_bf16_kernel(const float* __restrict__ in,
                                   u16* __restrict__ out, int n) {
    int i = blockIdx.x * blockDim.x + threadIdx.x;
    int stride = gridDim.x * blockDim.x;
    for (; i < n; i += stride) out[i] = f2bf(in[i]);
}

// ---------------------------------------------------------------------------
// Kernel 2: C = A * B^T + bias  (A:[M,K] bf16 rows, B:[N,K] bf16 rows)
// Block tile 256(M) x 64(N): 8 waves, each 32x64 (8 accumulators).
// B 64x32 chunk double-buffered in LDS via async copies (4KB = 256 thr x 16B).
// ---------------------------------------------------------------------------
enum { MODE_QK = 0, MODE_VT = 1, MODE_F32 = 2 };

template <int MODE>
__global__ __launch_bounds__(256)
void gemm_bt_kernel(const u16* __restrict__ A, const u16* __restrict__ Bw,
                    const float* __restrict__ bias, void* __restrict__ outp,
                    int M, int N, int K) {
    __shared__ __align__(16) u16 Bs[2][64 * 32];   // [n][k] bf16, 2 x 4KB

    const int tid  = threadIdx.x;
    const int w    = tid >> 5;
    const int lane = tid & 31;
    const int nl   = lane & 15;
    const int rb   = (lane < 16) ? 0 : 8;
    const int m0   = blockIdx.y * 256 + w * 32;
    const int n0   = blockIdx.x * 64;

    const u16* Arow0 = A + (size_t)(m0 + nl) * K;        // M sub-tile 0
    const u16* Arow1 = A + (size_t)(m0 + 16 + nl) * K;   // M sub-tile 1
    // per-thread async source: row n0 + tid/4, k chunk (tid%4)*8 (+kb)
    const u16* gB = Bw + (size_t)(n0 + (tid >> 2)) * K + (tid & 3) * 8;

    async_copy_b128(&Bs[0][tid * 8], gB);                // prologue: kb = 0

    v8f acc[8] = {};
    for (int kb = 0; kb < K; kb += 32) {
        const int buf = (kb >> 5) & 1;
        wait_async0();
        __syncthreads();
        if (kb + 32 < K)
            async_copy_b128(&Bs[buf ^ 1][tid * 8], gB + kb + 32);

        __builtin_prefetch(Arow0 + kb + 64, 0, 1);
        v16bf a0 = load_frag_A(Arow0 + kb, lane);
        v16bf a1 = load_frag_A(Arow1 + kb, lane);
        const u16* Bt = Bs[buf];
#pragma unroll
        for (int t = 0; t < 4; ++t) {
            v16bf bfr  = load_frag_B(Bt + (t * 16 + nl) * 32, lane);
            acc[t]     = wmma_bf16(a0, bfr, acc[t]);
            acc[4 + t] = wmma_bf16(a1, bfr, acc[4 + t]);
        }
    }

#pragma unroll
    for (int s = 0; s < 2; ++s) {
#pragma unroll
        for (int t = 0; t < 4; ++t) {
            const int gn = n0 + t * 16 + nl;
            const float bn = bias[gn];
#pragma unroll
            for (int i = 0; i < 8; ++i) {
                const int gm = m0 + s * 16 + rb + i;
                const float v = acc[s * 4 + t][i] + bn;
                if (MODE == MODE_F32) {
                    ((float*)outp)[(size_t)gm * N + gn] = v;
                } else {
                    const int b = gm / SS, ss = gm % SS;
                    const int h = gn / DK, d = gn % DK;
                    if (MODE == MODE_QK)
                        ((u16*)outp)[(((size_t)(b * HH + h) * SS) + ss) * DK + d] = f2bf(v);
                    else   // MODE_VT: V stored transposed [B,H,Dk,S]
                        ((u16*)outp)[(((size_t)(b * HH + h) * DK) + d) * SS + ss] = f2bf(v);
                }
            }
        }
    }
}

// ---------------------------------------------------------------------------
// Kernel 3: flash attention. 8 waves share one (b,h); K/V 32-key tiles are
// double-buffered in LDS via async copies (2 x 4KB each). Per wave: 16 q rows,
// 4 WMMAs QK^T, online softmax (ds_swizzle reductions), P relaid via LDS into
// A-fragment form, 4 WMMAs PV with rescaled accumulators.
// ---------------------------------------------------------------------------
__global__ __launch_bounds__(256)
void flash_attn_kernel(const u16* __restrict__ Qb, const u16* __restrict__ Kb,
                       const u16* __restrict__ Vtb, u16* __restrict__ ctx) {
    __shared__ __align__(16) u16 Ks[2][32 * 64];   // [key][dk]   2 x 4KB
    __shared__ __align__(16) u16 Vs[2][64 * 32];   // [dk][key]   2 x 4KB
    __shared__ __align__(16) u16 Pb[8][16 * 32];   // per-wave P  8KB

    const int tid  = threadIdx.x;
    const int w    = tid >> 5;
    const int lane = tid & 31;
    const int nl   = lane & 15;
    const int rb   = (lane < 16) ? 0 : 8;

    const int nqb  = SS / 128;
    const int qblk = blockIdx.x % nqb;
    const int bh   = blockIdx.x / nqb;
    const int b    = bh / HH;
    const int h    = bh % HH;
    const int q0   = qblk * 128 + w * 16;

    const u16* Qh = Qb  + (size_t)bh * SS * DK;
    const u16* Kh = Kb  + (size_t)bh * SS * DK;
    const u16* Vh = Vtb + (size_t)bh * DK * SS;

    const u16* qrow = Qh + (size_t)(q0 + nl) * DK;
    const v16bf qa0 = load_frag_A(qrow, lane);
    const v16bf qa1 = load_frag_A(qrow + 32, lane);

    // per-thread async staging sources (add kb*DK / kb per block)
    const u16* gK = Kh + (size_t)(tid >> 3) * DK + (tid & 7) * 8;  // key tid/8, dk chunk
    const u16* gV = Vh + (size_t)(tid >> 2) * SS + (tid & 3) * 8;  // dk row tid/4, key chunk

    async_copy_b128(&Ks[0][tid * 8], gK);     // kb = 0
    async_copy_b128(&Vs[0][tid * 8], gV);

    v8f o0 = {}, o1 = {}, o2 = {}, o3 = {};
    float mx[8], l[8];
#pragma unroll
    for (int i = 0; i < 8; ++i) { mx[i] = -1e30f; l[i] = 0.0f; }
    const float sc = 0.125f;   // 1/sqrt(64)
    u16* P = &Pb[w][0];

    for (int kb = 0; kb < SS; kb += 32) {
        const int buf = (kb >> 5) & 1;
        wait_async0();
        __syncthreads();
        if (kb + 32 < SS) {
            async_copy_b128(&Ks[buf ^ 1][tid * 8], gK + (size_t)(kb + 32) * DK);
            async_copy_b128(&Vs[buf ^ 1][tid * 8], gV + (kb + 32));
        }

        // ---- scores S[16 q x 32 keys] from LDS K tile ----
        const u16* Kt = Ks[buf];
        v8f s0 = {}, s1 = {};
        s0 = wmma_bf16(qa0, load_frag_B(Kt + nl * 64, lane), s0);
        s0 = wmma_bf16(qa1, load_frag_B(Kt + nl * 64 + 32, lane), s0);
        s1 = wmma_bf16(qa0, load_frag_B(Kt + (16 + nl) * 64, lane), s1);
        s1 = wmma_bf16(qa1, load_frag_B(Kt + (16 + nl) * 64 + 32, lane), s1);

        // ---- online softmax; row r=rb+i lives in vgpr i across 16 lanes ----
#pragma unroll
        for (int i = 0; i < 8; ++i) {
            float v0 = s0[i] * sc, v1 = s1[i] * sc;
            float bm  = red16_max(fmaxf(v0, v1));
            float mn  = fmaxf(mx[i], bm);
            float esc = __expf(mx[i] - mn);
            float p0  = __expf(v0 - mn);
            float p1  = __expf(v1 - mn);
            float rs  = red16_sum(p0 + p1);
            l[i]  = l[i] * esc + rs;
            mx[i] = mn;
            o0[i] *= esc; o1[i] *= esc; o2[i] *= esc; o3[i] *= esc;
            const int r = rb + i;
            P[r * 32 + nl]      = f2bf(p0);
            P[r * 32 + 16 + nl] = f2bf(p1);
        }
        asm volatile("s_wait_dscnt 0x0" ::: "memory");

        // ---- ctx += P * V from LDS V tile ----
        const v16bf pa = load_frag_A(P + nl * 32, lane);
        const u16* Vt = Vs[buf];
        o0 = wmma_bf16(pa, load_frag_B(Vt + ( 0 + nl) * 32, lane), o0);
        o1 = wmma_bf16(pa, load_frag_B(Vt + (16 + nl) * 32, lane), o1);
        o2 = wmma_bf16(pa, load_frag_B(Vt + (32 + nl) * 32, lane), o2);
        o3 = wmma_bf16(pa, load_frag_B(Vt + (48 + nl) * 32, lane), o3);
    }

    // ---- finalize: divide by row sums, store bf16 ctx as [B,S,H*Dk] ----
    v8f oo[4] = {o0, o1, o2, o3};
#pragma unroll
    for (int i = 0; i < 8; ++i) {
        const float inv = 1.0f / l[i];
        const size_t row = ((size_t)b * SS + q0 + rb + i) * DD + (size_t)h * DK;
#pragma unroll
        for (int t = 0; t < 4; ++t)
            ctx[row + t * 16 + nl] = f2bf(oo[t][i] * inv);
    }
}

// ---------------------------------------------------------------------------
// Host launcher
// ---------------------------------------------------------------------------
extern "C" void kernel_launch(void* const* d_in, const int* in_sizes, int n_in,
                              void* d_out, int out_size, void* d_ws, size_t ws_size,
                              hipStream_t stream) {
    const float* x  = (const float*)d_in[0];
    const float* Wq = (const float*)d_in[1];
    const float* bq = (const float*)d_in[2];
    const float* Wk = (const float*)d_in[3];
    const float* bk = (const float*)d_in[4];
    const float* Wv = (const float*)d_in[5];
    const float* bv = (const float*)d_in[6];
    const float* Wo = (const float*)d_in[7];
    const float* bo = (const float*)d_in[8];
    float* out = (float*)d_out;

    const size_t XE = (size_t)BB * SS * DD;   // 8,388,608
    const size_t WE = (size_t)DD * DD;        // 1,048,576

    u16* ws   = (u16*)d_ws;
    u16* xb   = ws;
    u16* wqb  = xb  + XE;
    u16* wkb  = wqb + WE;
    u16* wvb  = wkb + WE;
    u16* wob  = wvb + WE;
    u16* qb   = wob + WE;       // Q  [B,H,S,Dk]
    u16* kbuf = qb  + XE;       // K  [B,H,S,Dk]
    u16* vtb  = kbuf + XE;      // V^T [B,H,Dk,S]
    u16* ctxb = vtb + XE;       // ctx [B,S,D]

    f32_to_bf16_kernel<<<1024, 256, 0, stream>>>(x,  xb,  (int)XE);
    f32_to_bf16_kernel<<<1024, 256, 0, stream>>>(Wq, wqb, (int)WE);
    f32_to_bf16_kernel<<<1024, 256, 0, stream>>>(Wk, wkb, (int)WE);
    f32_to_bf16_kernel<<<1024, 256, 0, stream>>>(Wv, wvb, (int)WE);
    f32_to_bf16_kernel<<<1024, 256, 0, stream>>>(Wo, wob, (int)WE);

    dim3 gg(DD / 64, (BB * SS) / 256), blk(256);
    gemm_bt_kernel<MODE_QK><<<gg, blk, 0, stream>>>(xb, wqb, bq, qb,   BB * SS, DD, DD);
    gemm_bt_kernel<MODE_QK><<<gg, blk, 0, stream>>>(xb, wkb, bk, kbuf, BB * SS, DD, DD);
    gemm_bt_kernel<MODE_VT><<<gg, blk, 0, stream>>>(xb, wvb, bv, vtb,  BB * SS, DD, DD);

    flash_attn_kernel<<<BB * HH * (SS / 128), 256, 0, stream>>>(qb, kbuf, vtb, ctxb);

    gemm_bt_kernel<MODE_F32><<<gg, blk, 0, stream>>>(ctxb, wob, bo, out, BB * SS, DD, DD);
}
